// PMESPEncoder_51453708206753
// MI455X (gfx1250) — compile-verified, matrix operand order checked by reading, not run on hardware
//
#include <hip/hip_runtime.h>

// ---------------- constants ----------------
#define NN    3600
#define MP    3648      // GEMM M padded to multiple of 64
#define HH    512
#define PES   588
#define PROT  384
#define OCC   256
#define EE    230400
#define G4H   2048      // 4*H
#define K0P   1120      // pad(588+512, 32)
#define K1P   1536      // 1024+512 (already mult of 32)
#define KFP   1408      // PROT + 2*H (mult of 32)

typedef __bf16 bf16_t;
typedef bf16_t v16bf __attribute__((ext_vector_type(16)));
typedef float  v8f   __attribute__((ext_vector_type(8)));

union ABFrag { v16bf v; uint4 q[2]; };

__device__ __forceinline__ unsigned short f2bf(float f) {
    union { float f; unsigned int u; } x; x.f = f;
    unsigned int u = x.u;
    unsigned int r = u + 0x7FFFu + ((u >> 16) & 1u);   // round-to-nearest-even
    return (unsigned short)(r >> 16);
}
__device__ __forceinline__ float sigm(float v) { return 1.0f / (1.0f + __expf(-v)); }

// ---------------- fill / pack kernels ----------------
__global__ void k_fill_u16(unsigned short* p, unsigned short v, long n) {
    long i = (long)blockIdx.x * blockDim.x + threadIdx.x;
    if (i < n) p[i] = v;
}
__global__ void k_fill_f32(float* p, float v, int n) {
    int i = blockIdx.x * blockDim.x + threadIdx.x;
    if (i < n) p[i] = v;
}
// dst[r*ldd + dstOff + c] = bf16(src[r*lds + srcOff + c] * scale)
__global__ void k_pack(unsigned short* __restrict__ dst, int ldd, int dstOff,
                       const float* __restrict__ src, int lds, int srcOff,
                       int rows, int cols, float scale) {
    long i = (long)blockIdx.x * blockDim.x + threadIdx.x;
    if (i >= (long)rows * cols) return;
    int r = (int)(i / cols), c = (int)(i - (long)r * cols);
    dst[(size_t)r * ldd + dstOff + c] = f2bf(src[(size_t)r * lds + srcOff + c] * scale);
}
// dst[n*ldd + k] = bf16(src[k*srcLd + n])   (transpose: src is (K x Ncols))
__global__ void k_pack_t(unsigned short* __restrict__ dst, int ldd,
                         const float* __restrict__ src, int srcLd,
                         int dstRows, int dstCols) {
    long i = (long)blockIdx.x * blockDim.x + threadIdx.x;
    if (i >= (long)dstRows * dstCols) return;
    int n = (int)(i / dstCols), k = (int)(i - (long)n * dstCols);
    dst[(size_t)n * ldd + k] = f2bf(src[(size_t)k * srcLd + n]);
}

// ---------------- WMMA GEMM: C(MPxNc,f32) = A(MPxKp,bf16) * B(NcxKp,bf16)^T ----------------
// block = 256 threads = 8 waves arranged 2(M) x 4(N); each wave computes a 32x64 patch
// as a 2x4 grid of 16x16 WMMA tiles -> 6 fragment loads feed 8 WMMAs (1.5 loads/WMMA).
// Block covers 64(M) x 256(N).
__global__ __launch_bounds__(256) void k_gemm_wmma(
    float* __restrict__ C,
    const unsigned short* __restrict__ A,
    const unsigned short* __restrict__ B,
    int Nc, int Kp)
{
    const int lane = threadIdx.x & 31;
    const int wave = threadIdx.x >> 5;
    const int wm   = wave >> 2;          // 0..1
    const int wn   = wave & 3;           // 0..3
    const int hi   = lane >> 4;          // K-half selector within a 32-K step
    const int lo   = lane & 15;          // row (A) / col (B) within tile
    const int m0   = blockIdx.y * 64 + wm * 32;
    const int n0   = blockIdx.x * 256 + wn * 64;

    const unsigned short* aRow[2];
    const unsigned short* bRow[4];
#pragma unroll
    for (int i = 0; i < 2; ++i) aRow[i] = A + (size_t)(m0 + 16 * i + lo) * Kp;
#pragma unroll
    for (int j = 0; j < 4; ++j) bRow[j] = B + (size_t)(n0 + 16 * j + lo) * Kp;

    v8f acc[2][4];
#pragma unroll
    for (int i = 0; i < 2; ++i)
#pragma unroll
        for (int j = 0; j < 4; ++j)
#pragma unroll
            for (int r = 0; r < 8; ++r) acc[i][j][r] = 0.f;

    for (int k0 = 0; k0 < Kp; k0 += 32) {
        const int kb = k0 + hi * 8;
        ABFrag a[2], b[4];
        // ISA 7.12.2: 16-bit A 16x32 — lane(lo) holds row m, halves K=[hi*8,+8) and [16+hi*8,+8)
#pragma unroll
        for (int i = 0; i < 2; ++i) {
            a[i].q[0] = *reinterpret_cast<const uint4*>(aRow[i] + kb);
            a[i].q[1] = *reinterpret_cast<const uint4*>(aRow[i] + kb + 16);
        }
#pragma unroll
        for (int j = 0; j < 4; ++j) {
            b[j].q[0] = *reinterpret_cast<const uint4*>(bRow[j] + kb);
            b[j].q[1] = *reinterpret_cast<const uint4*>(bRow[j] + kb + 16);
        }
#pragma unroll
        for (int i = 0; i < 2; ++i)
#pragma unroll
            for (int j = 0; j < 4; ++j)
                acc[i][j] = __builtin_amdgcn_wmma_f32_16x16x32_bf16(
                    false, a[i].v, false, b[j].v, (short)0, acc[i][j], false, false);
    }

    // C/D layout: lanes 0-15 -> M=0..7 in vgpr 0..7; lanes 16-31 -> M=8..15
#pragma unroll
    for (int i = 0; i < 2; ++i) {
        float* cbase = C + (size_t)(m0 + 16 * i + hi * 8) * Nc + (n0 + lo);
#pragma unroll
        for (int j = 0; j < 4; ++j) {
            float* cp = cbase + 16 * j;
#pragma unroll
            for (int r = 0; r < 8; ++r) cp[(size_t)r * Nc] = acc[i][j][r];
        }
    }
}

// ---------------- LSTM pointwise ----------------
// g: (MP x 2048) gates pre-activation (i,f,g,o blocks of 512). Writes bf16 h into 1-2 dests.
__global__ void k_lstm(const float* __restrict__ g,
                       const float* __restrict__ bih, const float* __restrict__ bhh,
                       const float* __restrict__ c0,
                       unsigned short* __restrict__ dst1, int ld1, int off1,
                       unsigned short* __restrict__ dst2, int ld2, int off2,
                       float scale, int n) {
    int idx = blockIdx.x * blockDim.x + threadIdx.x;
    if (idx >= n) return;
    int node = idx >> 9;            // /512
    int h    = idx & 511;
    const float* gr = g + (size_t)node * G4H;
    float iv = gr[h]          + bih[h]          + bhh[h];
    float fv = gr[512 + h]    + bih[512 + h]    + bhh[512 + h];
    float gv = gr[1024 + h]   + bih[1024 + h]   + bhh[1024 + h];
    float ov = gr[1536 + h]   + bih[1536 + h]   + bhh[1536 + h];
    float c2 = sigm(fv) * c0[idx] + sigm(iv) * tanhf(gv);
    float hv = sigm(ov) * tanhf(c2) * scale;
    unsigned short b = f2bf(hv);
    dst1[(size_t)node * ld1 + off1 + h] = b;
    if (dst2) dst2[(size_t)node * ld2 + off2 + h] = b;
}

// ---------------- GCN helpers ----------------
__global__ void k_deg(float* __restrict__ deg, const int* __restrict__ colIdx, int nE) {
    int e = blockIdx.x * blockDim.x + threadIdx.x;
    if (e < nE) atomicAdd(&deg[colIdx[e]], 1.0f);
}
__global__ void k_norm(float* __restrict__ nrm, const float* __restrict__ deg,
                       const int* __restrict__ rowIdx, const int* __restrict__ colIdx, int nE) {
    int e = blockIdx.x * blockDim.x + threadIdx.x;
    if (e < nE) nrm[e] = rsqrtf(deg[rowIdx[e]]) * rsqrtf(deg[colIdx[e]]);
}
// out[node*F+f] = bias[f] + xw[node*F+f] / deg[node]   (bias + self-loop term)
__global__ void k_init_out(float* __restrict__ out, const float* __restrict__ xw,
                           const float* __restrict__ bias, const float* __restrict__ deg,
                           int F, int n) {
    int idx = blockIdx.x * blockDim.x + threadIdx.x;
    if (idx >= n) return;
    int node = idx / F, f = idx - node * F;
    out[idx] = bias[f] + xw[idx] / deg[node];
}
__global__ void k_scatter(float* __restrict__ out, const float* __restrict__ xw,
                          const int* __restrict__ rowIdx, const int* __restrict__ colIdx,
                          const float* __restrict__ nrm, int F, int nE) {
    int e = blockIdx.x;
    if (e >= nE) return;
    int r = rowIdx[e], c = colIdx[e];
    float w = nrm[e];
    const float* src = xw + (size_t)r * F;
    float*       dst = out + (size_t)c * F;
    for (int f = threadIdx.x; f < F; f += blockDim.x)
        atomicAdd(&dst[f], src[f] * w);
}
__global__ void k_relu_bf16(unsigned short* __restrict__ dst, const float* __restrict__ src, int n) {
    int i = blockIdx.x * blockDim.x + threadIdx.x;
    if (i < n) { float v = src[i]; dst[i] = f2bf(v > 0.f ? v : 0.f); }
}

// ---------------- launch ----------------
static inline int cdiv(long a, long b) { return (int)((a + b - 1) / b); }

extern "C" void kernel_launch(void* const* d_in, const int* in_sizes, int n_in,
                              void* d_out, int out_size, void* d_ws, size_t ws_size,
                              hipStream_t stream) {
    const float* x      = (const float*)d_in[0];
    const int*   ei     = (const int*)d_in[1];       // (2, E)
    const float* h0     = (const float*)d_in[2];     // (4, N, H)
    const float* c0     = (const float*)d_in[3];
    const float* wih0f  = (const float*)d_in[4];
    const float* whh0f  = (const float*)d_in[5];
    const float* bih0f  = (const float*)d_in[6];
    const float* bhh0f  = (const float*)d_in[7];
    const float* wih0b  = (const float*)d_in[8];
    const float* whh0b  = (const float*)d_in[9];
    const float* bih0b  = (const float*)d_in[10];
    const float* bhh0b  = (const float*)d_in[11];
    const float* wih1f  = (const float*)d_in[12];
    const float* whh1f  = (const float*)d_in[13];
    const float* bih1f  = (const float*)d_in[14];
    const float* bhh1f  = (const float*)d_in[15];
    const float* wih1b  = (const float*)d_in[16];
    const float* whh1b  = (const float*)d_in[17];
    const float* bih1b  = (const float*)d_in[18];
    const float* bhh1b  = (const float*)d_in[19];
    const float* W1     = (const float*)d_in[20];    // (1408, 512)
    const float* b1     = (const float*)d_in[21];
    const float* W2     = (const float*)d_in[22];    // (512, 256)
    const float* b2     = (const float*)d_in[23];
    float* out = (float*)d_out;

    const int* rows = ei;           // edge_index[0]
    const int* cols = ei + EE;      // edge_index[1]

    // ---- workspace carve ----
    char* ws = (char*)d_ws;
    size_t off = 0;
    auto carve = [&](size_t bytes) -> void* {
        void* p = ws + off;
        off = (off + bytes + 255) & ~(size_t)255;
        return p;
    };
    float*          deg   = (float*)carve(sizeof(float) * NN);
    float*          nrm   = (float*)carve(sizeof(float) * EE);
    unsigned short* A0f   = (unsigned short*)carve(2ul * MP * K0P);
    unsigned short* A0b   = (unsigned short*)carve(2ul * MP * K0P);
    unsigned short* W0f   = (unsigned short*)carve(2ul * G4H * K0P);
    unsigned short* W0b   = (unsigned short*)carve(2ul * G4H * K0P);
    unsigned short* A1f   = (unsigned short*)carve(2ul * MP * K1P);
    unsigned short* A1b   = (unsigned short*)carve(2ul * MP * K1P);
    unsigned short* Wl1f  = (unsigned short*)carve(2ul * G4H * K1P);
    unsigned short* Wl1b  = (unsigned short*)carve(2ul * G4H * K1P);
    unsigned short* featA = (unsigned short*)carve(2ul * MP * KFP);
    unsigned short* W1T   = (unsigned short*)carve(2ul * HH * KFP);     // (512 x 1408)
    unsigned short* W2T   = (unsigned short*)carve(2ul * OCC * HH);     // (256 x 512)
    float*          gbuf  = (float*)carve(4ul * MP * G4H);
    float*          xw1   = (float*)carve(4ul * MP * HH);
    float*          out1  = (float*)carve(4ul * NN * HH);
    unsigned short* h1bf  = (unsigned short*)carve(2ul * MP * HH);
    float*          xw2   = (float*)carve(4ul * MP * OCC);
    (void)ws_size; (void)in_sizes; (void)n_in; (void)out_size;

    const int T = 256;
#define GRID1(n) cdiv((long)(n), T), T, 0, stream

    // ---- zero padded operand buffers / tails so WMMA inputs stay finite ----
    k_fill_u16<<<GRID1((long)MP * K0P)>>>(A0f, 0, (long)MP * K0P);
    k_fill_u16<<<GRID1((long)MP * K0P)>>>(A0b, 0, (long)MP * K0P);
    k_fill_u16<<<GRID1((long)G4H * K0P)>>>(W0f, 0, (long)G4H * K0P);
    k_fill_u16<<<GRID1((long)G4H * K0P)>>>(W0b, 0, (long)G4H * K0P);
    k_fill_u16<<<GRID1((long)(MP - NN) * K1P)>>>(A1f + (size_t)NN * K1P, 0, (long)(MP - NN) * K1P);
    k_fill_u16<<<GRID1((long)(MP - NN) * K1P)>>>(A1b + (size_t)NN * K1P, 0, (long)(MP - NN) * K1P);
    k_fill_u16<<<GRID1((long)(MP - NN) * KFP)>>>(featA + (size_t)NN * KFP, 0, (long)(MP - NN) * KFP);
    k_fill_u16<<<GRID1((long)(MP - NN) * HH)>>>(h1bf + (size_t)NN * HH, 0, (long)(MP - NN) * HH);

    // ---- pack operands to bf16 ----
    // A0 = [pes | h0[dir]]
    k_pack<<<GRID1((long)NN * PES)>>>(A0f, K0P, 0,   x,  PES + PROT, 0, NN, PES, 1.f);
    k_pack<<<GRID1((long)NN * HH)>>>(A0f, K0P, PES, h0 + 0ul * NN * HH, HH, 0, NN, HH, 1.f);
    k_pack<<<GRID1((long)NN * PES)>>>(A0b, K0P, 0,   x,  PES + PROT, 0, NN, PES, 1.f);
    k_pack<<<GRID1((long)NN * HH)>>>(A0b, K0P, PES, h0 + 1ul * NN * HH, HH, 0, NN, HH, 1.f);
    // W0 = [wih | whh]
    k_pack<<<GRID1((long)G4H * PES)>>>(W0f, K0P, 0,   wih0f, PES, 0, G4H, PES, 1.f);
    k_pack<<<GRID1((long)G4H * HH)>>>(W0f, K0P, PES, whh0f, HH,  0, G4H, HH, 1.f);
    k_pack<<<GRID1((long)G4H * PES)>>>(W0b, K0P, 0,   wih0b, PES, 0, G4H, PES, 1.f);
    k_pack<<<GRID1((long)G4H * HH)>>>(W0b, K0P, PES, whh0b, HH,  0, G4H, HH, 1.f);
    // layer-1 weights [wih(2048x1024) | whh(2048x512)]
    k_pack<<<GRID1((long)G4H * 1024)>>>(Wl1f, K1P, 0,    wih1f, 1024, 0, G4H, 1024, 1.f);
    k_pack<<<GRID1((long)G4H * HH)>>>(Wl1f, K1P, 1024, whh1f, HH,   0, G4H, HH, 1.f);
    k_pack<<<GRID1((long)G4H * 1024)>>>(Wl1b, K1P, 0,    wih1b, 1024, 0, G4H, 1024, 1.f);
    k_pack<<<GRID1((long)G4H * HH)>>>(Wl1b, K1P, 1024, whh1b, HH,   0, G4H, HH, 1.f);
    // A1 h-portion (cols 1024..1535) = h0[2] / h0[3]; x1 part filled by k_lstm later
    k_pack<<<GRID1((long)NN * HH)>>>(A1f, K1P, 1024, h0 + 2ul * NN * HH, HH, 0, NN, HH, 1.f);
    k_pack<<<GRID1((long)NN * HH)>>>(A1b, K1P, 1024, h0 + 3ul * NN * HH, HH, 0, NN, HH, 1.f);
    // feat cols 0..383 = prot
    k_pack<<<GRID1((long)NN * PROT)>>>(featA, KFP, 0, x, PES + PROT, PES, NN, PROT, 1.f);
    // transposed GCN weights
    k_pack_t<<<GRID1((long)HH * KFP)>>>(W1T, KFP, W1, HH, HH, KFP);
    k_pack_t<<<GRID1((long)OCC * HH)>>>(W2T, HH, W2, OCC, OCC, HH);

    // ---- degrees / norms (self loop => deg init 1.0) ----
    k_fill_f32<<<GRID1(NN)>>>(deg, 1.0f, NN);
    k_deg<<<GRID1(EE)>>>(deg, cols, EE);
    k_norm<<<GRID1(EE)>>>(nrm, deg, rows, cols, EE);

    // ---- LSTM layer 0 (fwd, bwd) ----
    dim3 blk(256);
    dim3 gLstm(G4H / 256, MP / 64);
    k_gemm_wmma<<<gLstm, blk, 0, stream>>>(gbuf, A0f, W0f, G4H, K0P);
    k_lstm<<<GRID1((long)NN * HH)>>>(gbuf, bih0f, bhh0f, c0 + 0ul * NN * HH,
                                     A1f, K1P, 0, A1b, K1P, 0, 1.0f, NN * HH);
    k_gemm_wmma<<<gLstm, blk, 0, stream>>>(gbuf, A0b, W0b, G4H, K0P);
    k_lstm<<<GRID1((long)NN * HH)>>>(gbuf, bih0b, bhh0b, c0 + 1ul * NN * HH,
                                     A1f, K1P, HH, A1b, K1P, HH, 1.0f, NN * HH);

    // ---- LSTM layer 1 (fwd, bwd) -> feat cols (scaled by 1/100) ----
    k_gemm_wmma<<<gLstm, blk, 0, stream>>>(gbuf, A1f, Wl1f, G4H, K1P);
    k_lstm<<<GRID1((long)NN * HH)>>>(gbuf, bih1f, bhh1f, c0 + 2ul * NN * HH,
                                     featA, KFP, PROT, (unsigned short*)nullptr, 0, 0,
                                     0.01f, NN * HH);
    k_gemm_wmma<<<gLstm, blk, 0, stream>>>(gbuf, A1b, Wl1b, G4H, K1P);
    k_lstm<<<GRID1((long)NN * HH)>>>(gbuf, bih1b, bhh1b, c0 + 3ul * NN * HH,
                                     featA, KFP, PROT + HH, (unsigned short*)nullptr, 0, 0,
                                     0.01f, NN * HH);

    // ---- GCN layer 1: xw1 = feat @ W1 ----
    dim3 gG1(HH / 256, MP / 64);
    k_gemm_wmma<<<gG1, blk, 0, stream>>>(xw1, featA, W1T, HH, KFP);
    k_init_out<<<GRID1((long)NN * HH)>>>(out1, xw1, b1, deg, HH, NN * HH);
    k_scatter<<<EE, 128, 0, stream>>>(out1, xw1, rows, cols, nrm, HH, EE);
    k_relu_bf16<<<GRID1((long)NN * HH)>>>(h1bf, out1, NN * HH);

    // ---- GCN layer 2: xw2 = h1 @ W2 ----
    dim3 gG2(OCC / 256, MP / 64);
    k_gemm_wmma<<<gG2, blk, 0, stream>>>(xw2, h1bf, W2T, OCC, HH);
    k_init_out<<<GRID1((long)NN * OCC)>>>(out, xw2, b2, deg, OCC, NN * OCC);
    k_scatter<<<EE, 128, 0, stream>>>(out, xw2, rows, cols, nrm, OCC, EE);
#undef GRID1
}
